// MHA_80633716015579
// MI455X (gfx1250) — compile-verified, matrix operand order checked by reading, not run on hardware
//
#include <hip/hip_runtime.h>
#include <hip/hip_bf16.h>

// ---------------------------------------------------------------------------
// MI455X (gfx1250) attention layer.
//  - All heavy math through v_wmma_f32_16x16x32_bf16 (wave32 WMMA).
//  - Tiles staged global->LDS with GLOBAL_LOAD_ASYNC_TO_LDS_B128 (ASYNCcnt)
//    via __builtin_amdgcn_global_load_async_to_lds_b128.
//  - Double-buffered 128x128 GEMM, 8 WMMAs per wave per 32-K step.
//  - Flash-attention style online softmax; V stored transposed so the PV
//    B-operand is a contiguous ds_load_b128 in the native WMMA B layout.
// ---------------------------------------------------------------------------

#define SEQ       2048
#define DIM       4096
#define NH        32
#define NKV       8
#define HD        128
#define KV_DIM    (NKV * HD)   // 1024

typedef __attribute__((ext_vector_type(16))) __bf16 v16bf;
typedef __attribute__((ext_vector_type(8)))  __bf16 v8bf;
typedef __attribute__((ext_vector_type(8)))  float  v8f;
typedef __attribute__((ext_vector_type(4)))  int    v4i;

#if __has_builtin(__builtin_amdgcn_global_load_async_to_lds_b128)
#define HAS_ASYNC_COPY 1
// pointee types with explicit address spaces (AS1 = global, AS3 = LDS)
typedef __attribute__((address_space(1))) v4i v4i_glob;
typedef __attribute__((address_space(3))) v4i v4i_lds;
#else
#define HAS_ASYNC_COPY 0
#endif

// 16-byte global -> LDS copy (per lane).
__device__ __forceinline__ void copy16_g2l(const __bf16* g, __bf16* l) {
#if HAS_ASYNC_COPY
    __builtin_amdgcn_global_load_async_to_lds_b128(
        (v4i_glob*)(unsigned long long)g,
        (v4i_lds*)(unsigned long long)l,
        0, 0);
#else
    *(v8bf*)l = *(const v8bf*)g;
#endif
}

__device__ __forceinline__ void async_drain() {
#if HAS_ASYNC_COPY
#if __has_builtin(__builtin_amdgcn_s_wait_asynccnt)
    __builtin_amdgcn_s_wait_asynccnt(0);
#else
    asm volatile("s_wait_asynccnt 0" ::: "memory");
#endif
#endif
}

// Build a 16-element bf16 A/B fragment from two contiguous 16-byte chunks.
// Layout (ISA 7.12.2, 16-bit 16x32): lanes 0-15 hold K={0-7,16-23},
// lanes 16-31 hold K={8-15,24-31}; caller passes half = lane>>4.
__device__ __forceinline__ v16bf load_frag(const __bf16* row, int half) {
    v8bf lo = *(const v8bf*)(row + half * 8);
    v8bf hi = *(const v8bf*)(row + 16 + half * 8);
    return __builtin_shufflevector(lo, hi, 0,1,2,3,4,5,6,7,8,9,10,11,12,13,14,15);
}

__device__ __forceinline__ v8f wmma_bf16(v16bf a, v16bf b, v8f c) {
    return __builtin_amdgcn_wmma_f32_16x16x32_bf16(
        /*neg_a=*/false, a, /*neg_b=*/false, b,
        /*c_mod=*/(short)0, c, /*reuse_a=*/false, /*reuse_b=*/false);
}

// ---------------------------------------------------------------------------
// fp32 -> bf16 elementwise
// ---------------------------------------------------------------------------
__global__ void cvt_bf16_kernel(const float* __restrict__ in,
                                __bf16* __restrict__ out, int n) {
    int i = blockIdx.x * blockDim.x + threadIdx.x;
    if (i < n) out[i] = (__bf16)in[i];
}

// ---------------------------------------------------------------------------
// C(MxN,f32) = A(MxK,bf16 row-major) * B(NxK,bf16 row-major)^T
// Block tile 128x128, K-step 32, double-buffered LDS (async staging).
// 8 waves: waveM in 0..3 (32 rows), waveN in 0..1 (64 cols);
// per wave 2x4 = 8 WMMAs per K-step with A-frag reuse.
// M,N,K multiples of 128/32 here (no tails needed).
// ---------------------------------------------------------------------------
#define GBM 128
#define GBN 128
#define GBK 32
#define GLDK 40   // padded K stride (80B rows, 16B-aligned)

__global__ __launch_bounds__(256) void gemm_bf16_kernel(
    const __bf16* __restrict__ A, const __bf16* __restrict__ B,
    float* __restrict__ C, int M, int N, int K) {
    __shared__ __bf16 As[2][GBM][GLDK];
    __shared__ __bf16 Bs[2][GBN][GLDK];

    const int t     = threadIdx.x;
    const int w     = t >> 5;
    const int lane  = t & 31;
    const int half  = lane >> 4;
    const int mrow  = lane & 15;
    const int waveM = w >> 1;   // 0..3 -> 32 rows each
    const int waveN = w & 1;    // 0..1 -> 64 cols each
    const int bM = blockIdx.y * GBM;
    const int bN = blockIdx.x * GBN;

    // cooperative tile staging: 128x32 bf16 per matrix, 256 thr * 2 x 16B
    const int lrow  = t >> 1;          // 0..127
    const int lcol0 = (t & 1) * 16;    // 0 or 16
    const __bf16* Ag = A + (size_t)(bM + lrow) * K + lcol0;
    const __bf16* Bg = B + (size_t)(bN + lrow) * K + lcol0;

    v8f acc[2][4];
#pragma unroll
    for (int tm = 0; tm < 2; ++tm)
#pragma unroll
        for (int tn = 0; tn < 4; ++tn) acc[tm][tn] = (v8f){};

    auto stage = [&](int buf, int k0) {
        copy16_g2l(Ag + k0,     &As[buf][lrow][lcol0]);
        copy16_g2l(Ag + k0 + 8, &As[buf][lrow][lcol0 + 8]);
        copy16_g2l(Bg + k0,     &Bs[buf][lrow][lcol0]);
        copy16_g2l(Bg + k0 + 8, &Bs[buf][lrow][lcol0 + 8]);
    };

    // prologue: tile 0 into buffer 0
    stage(0, 0);
    async_drain();
    __syncthreads();

    for (int k0 = 0; k0 < K; k0 += GBK) {
        const int buf = (k0 / GBK) & 1;
        if (k0 + GBK < K) {
            stage(buf ^ 1, k0 + GBK);
            __builtin_prefetch(Ag + k0 + 2 * GBK, 0, 3);  // global_prefetch_b8
        }

        v16bf af[2];
#pragma unroll
        for (int tm = 0; tm < 2; ++tm)
            af[tm] = load_frag(&As[buf][waveM * 32 + tm * 16 + mrow][0], half);
#pragma unroll
        for (int tn = 0; tn < 4; ++tn) {
            v16bf b = load_frag(&Bs[buf][waveN * 64 + tn * 16 + mrow][0], half);
#pragma unroll
            for (int tm = 0; tm < 2; ++tm)
                acc[tm][tn] = wmma_bf16(af[tm], b, acc[tm][tn]);
        }

        async_drain();      // our async loads into buf^1 are done
        __syncthreads();    // everyone done loading buf^1 / reading buf
    }

    // C layout: lane holds col n = mrow; VGPR r holds row m = r + 8*half
#pragma unroll
    for (int tm = 0; tm < 2; ++tm) {
        const int gm = bM + waveM * 32 + tm * 16 + 8 * half;
#pragma unroll
        for (int tn = 0; tn < 4; ++tn) {
            const int gn = bN + waveN * 64 + tn * 16 + mrow;
#pragma unroll
            for (int r = 0; r < 8; ++r)
                C[(size_t)(gm + r) * N + gn] = acc[tm][tn][r];
        }
    }
}

// ---------------------------------------------------------------------------
// RoPE + bf16 repack: in [SEQ][nHeads*128] f32 -> out [nHeads][SEQ][128] bf16
// ---------------------------------------------------------------------------
__global__ void rope_cvt_kernel(const float* __restrict__ in,
                                __bf16* __restrict__ out, int nHeads) {
    int idx = blockIdx.x * blockDim.x + threadIdx.x;
    int total = SEQ * nHeads * (HD / 2);
    if (idx >= total) return;
    int i = idx & 63;                 // pair index 0..63
    int h = (idx >> 6) % nHeads;
    int q = idx / (64 * nHeads);
    float freq = __powf(500000.0f, -(float)(2 * i) / (float)HD);
    float s, c;
    __sincosf((float)q * freq, &s, &c);
    const float* p = in + (size_t)q * (nHeads * HD) + h * HD + 2 * i;
    float e = p[0], o = p[1];
    __bf16* po = out + ((size_t)h * SEQ + q) * HD + 2 * i;
    po[0] = (__bf16)(e * c - o * s);
    po[1] = (__bf16)(e * s + o * c);
}

// ---------------------------------------------------------------------------
// V repack transposed: in [SEQ][KV_DIM] f32 -> out [NKV][HD][SEQ] bf16
// ---------------------------------------------------------------------------
__global__ void vtrans_kernel(const float* __restrict__ in,
                              __bf16* __restrict__ out) {
    int idx = blockIdx.x * blockDim.x + threadIdx.x;
    if (idx >= SEQ * KV_DIM) return;
    int d = idx & (KV_DIM - 1);
    int q = idx >> 10;
    int h  = d >> 7;
    int dd = d & 127;
    out[((size_t)h * HD + dd) * SEQ + q] = (__bf16)in[(size_t)q * KV_DIM + d];
}

// ---------------------------------------------------------------------------
// Flash attention (causal, GQA rep=4).
// Grid: (SEQ/128, NH). Block: 256 thr = 8 waves; wave w owns 16 queries.
// 64-key steps staged in LDS (async): Ks [key][dim], Vs transposed [dim][key].
// Scores: 16 WMMAs/step; PV: 16 WMMAs/step.
// P (C-layout f32) -> bf16 A-layout via per-wave LDS staging.
// ---------------------------------------------------------------------------
__global__ __launch_bounds__(256) void attn_kernel(
    const __bf16* __restrict__ Qb,   // [NH][SEQ][HD]
    const __bf16* __restrict__ Kb,   // [NKV][SEQ][HD]
    const __bf16* __restrict__ Vt,   // [NKV][HD][SEQ]
    __bf16* __restrict__ Out) {      // [SEQ][NH*HD]
    __shared__ __bf16 Ks[64][HD + 8];     // 272B rows (16B aligned)
    __shared__ __bf16 Vs[HD][72];         // 144B rows
    __shared__ __bf16 Ps[8][16][72];      // per-wave P staging

    const int t    = threadIdx.x;
    const int w    = t >> 5;
    const int lane = t & 31;
    const int half = lane >> 4;
    const int mrow = lane & 15;
    const int h    = blockIdx.y;
    const int hkv  = h >> 2;                       // GQA: 4 Q heads per KV head
    const int qTile = blockIdx.x * 128;
    const int qBaseW = qTile + w * 16;

    // Q fragments (16 queries x 128 dims = 4 A-frags), loaded once.
    const __bf16* qrow = Qb + ((size_t)h * SEQ + qBaseW + mrow) * HD;
    v16bf qf[4];
#pragma unroll
    for (int dc = 0; dc < 4; ++dc) qf[dc] = load_frag(qrow + dc * 32, half);

    v8f o[8];
#pragma unroll
    for (int d = 0; d < 8; ++d) o[d] = (v8f){};
    float rowmax[8], rowsum[8];
#pragma unroll
    for (int r = 0; r < 8; ++r) { rowmax[r] = -1e30f; rowsum[r] = 0.f; }

    const float scale = 0.08838834764831845f;      // 1/sqrt(128)
    const int kSteps = 2 * blockIdx.x + 2;         // keys 0 .. qTile+127

    for (int ks = 0; ks < kSteps; ++ks) {
        const int kBase = ks * 64;
        __syncthreads();
        {   // Ks: 64 keys x 128 dims ; 4 thr/row, 64B each
            int kk = t >> 2, c0 = (t & 3) * 32;
            const __bf16* src = Kb + ((size_t)hkv * SEQ + kBase + kk) * HD + c0;
#pragma unroll
            for (int j = 0; j < 4; ++j)
                copy16_g2l(src + j * 8, &Ks[kk][c0 + j * 8]);
        }
        {   // Vs: 128 dims x 64 keys ; 2 thr/row, 64B each
            int d = t >> 1, c0 = (t & 1) * 32;
            const __bf16* src = Vt + ((size_t)hkv * HD + d) * SEQ + kBase + c0;
#pragma unroll
            for (int j = 0; j < 4; ++j)
                copy16_g2l(src + j * 8, &Vs[d][c0 + j * 8]);
        }
        async_drain();
        __syncthreads();

        // ---- scores: 4 key-chunks x 4 dim-chunks ----
        v8f sc[4];
#pragma unroll
        for (int c = 0; c < 4; ++c) {
            v8f s = (v8f){};
#pragma unroll
            for (int dc = 0; dc < 4; ++dc) {
                v16bf b = load_frag(&Ks[c * 16 + mrow][dc * 32], half);
                s = wmma_bf16(qf[dc], b, s);
            }
            sc[c] = s;
        }

        // ---- scale + causal mask (lane col = key, VGPR r = query row) ----
#pragma unroll
        for (int c = 0; c < 4; ++c) {
            int kg = kBase + c * 16 + mrow;
#pragma unroll
            for (int r = 0; r < 8; ++r) {
                int qg = qBaseW + r + 8 * half;
                sc[c][r] = (kg <= qg) ? sc[c][r] * scale : -1e30f;
            }
        }

        // ---- online softmax ----
        float cmax[8];
#pragma unroll
        for (int r = 0; r < 8; ++r)
            cmax[r] = fmaxf(fmaxf(sc[0][r], sc[1][r]), fmaxf(sc[2][r], sc[3][r]));
#pragma unroll
        for (int m = 1; m < 16; m <<= 1)
#pragma unroll
            for (int r = 0; r < 8; ++r)
                cmax[r] = fmaxf(cmax[r], __shfl_xor(cmax[r], m, 32));

        float fac[8];
#pragma unroll
        for (int r = 0; r < 8; ++r) {
            float nm = fmaxf(rowmax[r], cmax[r]);
            fac[r]   = __expf(rowmax[r] - nm);
            rowmax[r] = nm;
        }

        float psum[8];
#pragma unroll
        for (int r = 0; r < 8; ++r) psum[r] = 0.f;
#pragma unroll
        for (int c = 0; c < 4; ++c)
#pragma unroll
            for (int r = 0; r < 8; ++r) {
                float p = __expf(sc[c][r] - rowmax[r]);
                psum[r] += p;
                Ps[w][r + 8 * half][c * 16 + mrow] = (__bf16)p;
            }
#pragma unroll
        for (int m = 1; m < 16; m <<= 1)
#pragma unroll
            for (int r = 0; r < 8; ++r)
                psum[r] += __shfl_xor(psum[r], m, 32);
#pragma unroll
        for (int r = 0; r < 8; ++r) rowsum[r] = rowsum[r] * fac[r] + psum[r];

        // rescale existing accumulators
#pragma unroll
        for (int d = 0; d < 8; ++d)
#pragma unroll
            for (int r = 0; r < 8; ++r) o[d][r] *= fac[r];

        // same-wave LDS write->read of Ps: LDS is in-order per wave, but keep
        // the compiler honest and drain DS before the transposed re-read.
        asm volatile("s_wait_dscnt 0" ::: "memory");

        // ---- P(16x64) . V(64x128) ----
#pragma unroll
        for (int kc = 0; kc < 2; ++kc) {
            v16bf a = load_frag(&Ps[w][mrow][kc * 32], half);
#pragma unroll
            for (int d = 0; d < 8; ++d) {
                v16bf b = load_frag(&Vs[d * 16 + mrow][kc * 32], half);
                o[d] = wmma_bf16(a, b, o[d]);
            }
        }
    }

    // ---- epilogue: normalize and store bf16 [q][h*128 + d] ----
#pragma unroll
    for (int d = 0; d < 8; ++d)
#pragma unroll
        for (int r = 0; r < 8; ++r) {
            int q = qBaseW + r + 8 * half;
            float v = o[d][r] / rowsum[r];
            Out[(size_t)q * DIM + h * HD + d * 16 + mrow] = (__bf16)v;
        }
}

// ---------------------------------------------------------------------------
// host-side orchestration
// ---------------------------------------------------------------------------
extern "C" void kernel_launch(void* const* d_in, const int* in_sizes, int n_in,
                              void* d_out, int out_size, void* d_ws, size_t ws_size,
                              hipStream_t stream) {
    (void)in_sizes; (void)n_in; (void)out_size; (void)ws_size;
    const float* x  = (const float*)d_in[0];
    const float* wq = (const float*)d_in[1];
    const float* wk = (const float*)d_in[2];
    const float* wv = (const float*)d_in[3];
    const float* wo = (const float*)d_in[4];
    float* out = (float*)d_out;

    char* ws = (char*)d_ws;
    size_t off = 0;
    auto carve = [&](size_t bytes) -> char* {
        char* p = ws + off;
        off += (bytes + 255) & ~(size_t)255;
        return p;
    };
    __bf16* xb   = (__bf16*)carve((size_t)SEQ * DIM * 2);
    __bf16* wqb  = (__bf16*)carve((size_t)DIM * DIM * 2);
    __bf16* wkb  = (__bf16*)carve((size_t)KV_DIM * DIM * 2);
    __bf16* wvb  = (__bf16*)carve((size_t)KV_DIM * DIM * 2);
    __bf16* wob  = (__bf16*)carve((size_t)DIM * DIM * 2);
    float*  Qf   = (float*) carve((size_t)SEQ * DIM * 4);
    float*  Kf   = (float*) carve((size_t)SEQ * KV_DIM * 4);
    float*  Vf   = (float*) carve((size_t)SEQ * KV_DIM * 4);
    __bf16* Qb   = (__bf16*)carve((size_t)SEQ * DIM * 2);
    __bf16* Kbb  = (__bf16*)carve((size_t)SEQ * KV_DIM * 2);
    __bf16* Vtb  = (__bf16*)carve((size_t)SEQ * KV_DIM * 2);
    __bf16* attn = (__bf16*)carve((size_t)SEQ * DIM * 2);

    const int TPB = 256;
    auto blocks = [](size_t n) { return (unsigned)((n + 255) / 256); };

    // 1) fp32 -> bf16
    cvt_bf16_kernel<<<blocks((size_t)SEQ * DIM), TPB, 0, stream>>>(x, xb, SEQ * DIM);
    cvt_bf16_kernel<<<blocks((size_t)DIM * DIM), TPB, 0, stream>>>(wq, wqb, DIM * DIM);
    cvt_bf16_kernel<<<blocks((size_t)KV_DIM * DIM), TPB, 0, stream>>>(wk, wkb, KV_DIM * DIM);
    cvt_bf16_kernel<<<blocks((size_t)KV_DIM * DIM), TPB, 0, stream>>>(wv, wvb, KV_DIM * DIM);
    cvt_bf16_kernel<<<blocks((size_t)DIM * DIM), TPB, 0, stream>>>(wo, wob, DIM * DIM);

    // 2) projections: Q = x wq^T, K = x wk^T, V = x wv^T
    gemm_bf16_kernel<<<dim3(DIM / GBN, SEQ / GBM), TPB, 0, stream>>>(xb, wqb, Qf, SEQ, DIM, DIM);
    gemm_bf16_kernel<<<dim3(KV_DIM / GBN, SEQ / GBM), TPB, 0, stream>>>(xb, wkb, Kf, SEQ, KV_DIM, DIM);
    gemm_bf16_kernel<<<dim3(KV_DIM / GBN, SEQ / GBM), TPB, 0, stream>>>(xb, wvb, Vf, SEQ, KV_DIM, DIM);

    // 3) RoPE + repack (Q,K), transpose-repack V
    rope_cvt_kernel<<<blocks((size_t)SEQ * NH * 64), TPB, 0, stream>>>(Qf, Qb, NH);
    rope_cvt_kernel<<<blocks((size_t)SEQ * NKV * 64), TPB, 0, stream>>>(Kf, Kbb, NKV);
    vtrans_kernel<<<blocks((size_t)SEQ * KV_DIM), TPB, 0, stream>>>(Vf, Vtb);

    // 4) causal flash attention
    attn_kernel<<<dim3(SEQ / 128, NH), TPB, 0, stream>>>(Qb, Kbb, Vtb, attn);

    // 5) output projection: out = attn wo^T
    gemm_bf16_kernel<<<dim3(DIM / GBN, SEQ / GBM), TPB, 0, stream>>>(attn, wob, out, SEQ, DIM, DIM);
}
